// CustomLSTMModel_31542239822014
// MI455X (gfx1250) — matrix operand order repaired and checked by reading, not compiled, not run on hardware
//
#include <hip/hip_runtime.h>
#include <hip/hip_bf16.h>

#define TT   256
#define BB   1024
#define EMBD 100
#define EPAD 128   // f16 embedding row padded to 128
#define H1   64
#define H2   32
#define NG1  256   // 4*H1
#define NG2  128   // 4*H2
#define NVOC 5000

typedef _Float16 v16h  __attribute__((ext_vector_type(16)));
typedef float    v8f   __attribute__((ext_vector_type(8)));
typedef _Float16 half8 __attribute__((ext_vector_type(8)));

__device__ __forceinline__ float sigm(float x)  { return 1.0f / (1.0f + __expf(-x)); }
__device__ __forceinline__ float tanha(float x) { return 2.0f / (1.0f + __expf(-2.0f * x)) - 1.0f; }
__device__ __forceinline__ v8f   splat8(float v){ v8f r = {v, v, v, v, v, v, v, v}; return r; }

__device__ __forceinline__ v16h combine16(half8 q0, half8 q1) {
    v16h a;
    #pragma unroll
    for (int e = 0; e < 8; ++e) { a[e] = q0[e]; a[8 + e] = q1[e]; }
    return a;
}

#define WMMA(A, B, C) \
    __builtin_amdgcn_wmma_f32_16x16x32_f16(false, (A), false, (B), (short)0, (C), false, false)

// ----------------------------------------------------------------------------
// Prologue: f32 emb [5000][100] -> f16 embh [5000][128], zero padded.
// ----------------------------------------------------------------------------
__global__ __launch_bounds__(256) void embcvt_kernel(
    const float* __restrict__ emb, _Float16* __restrict__ embh)
{
    int idx = blockIdx.x * 256 + threadIdx.x;
    if (idx >= NVOC * EPAD) return;
    int row = idx >> 7, k = idx & 127;
    embh[idx] = (k < EMBD) ? (_Float16)emb[row * EMBD + k] : (_Float16)0.0f;
}

// ----------------------------------------------------------------------------
// Layer 1: persistent LSTM. Grid 16 x 256 threads (8 waves).
// Wave pair (g, sub): 16 batch rows, sub owns gate-col half (ht = 2*sub+hl).
// h state exchanged via ping-pong LDS buffer, one __syncthreads per step.
// ----------------------------------------------------------------------------
__global__ __launch_bounds__(256) void lstm1_kernel(
    const int* __restrict__ tokens, const _Float16* __restrict__ embh,
    const float* __restrict__ W1, const float* __restrict__ U1,
    const float* __restrict__ b1, _Float16* __restrict__ hs1)
{
    __shared__ _Float16 Wf[4 * 16 * 512];     // 64 KB: 4 ktiles x 16 ntiles x frag
    __shared__ _Float16 Uf[2 * 16 * 512];     // 32 KB
    __shared__ _Float16 hbuf[2][4][16][H1];   // ping-pong h staging per row group
    __shared__ int      tkbuf[2][4][16];      // ping-pong token staging

    const int tid = threadIdx.x;
    const int wv  = tid >> 5;
    const int ln  = tid & 31;
    const int r   = ln & 15;
    const int hi  = ln >> 4;
    const int g   = wv >> 1;    // row group 0..3
    const int sub = wv & 1;     // gate-column half

    // ---- build pre-swizzled f16 B-matrix fragment blobs (once) ----
    for (int idx = tid; idx < 4 * 16 * 512; idx += 256) {
        int kt = idx >> 13;
        int nt = (idx >> 9) & 15;
        int q  = idx & 511;
        int fl = q >> 4, e = q & 15;
        int fr = fl & 15, fh = fl >> 4;
        int k = kt * 32 + fh * 8 + (e & 7) + (e >> 3) * 16;
        int n = nt * 16 + fr;
        Wf[idx] = (k < EMBD) ? (_Float16)W1[k * NG1 + n] : (_Float16)0.0f;
    }
    for (int idx = tid; idx < 2 * 16 * 512; idx += 256) {
        int kt = idx >> 13;
        int nt = (idx >> 9) & 15;
        int q  = idx & 511;
        int fl = q >> 4, e = q & 15;
        int fr = fl & 15, fh = fl >> 4;
        int k = kt * 32 + fh * 8 + (e & 7) + (e >> 3) * 16;
        int n = nt * 16 + fr;
        Uf[idx] = (_Float16)U1[k * NG1 + n];
    }
    for (int idx = tid; idx < 2 * 4 * 16 * H1; idx += 256)
        (&hbuf[0][0][0][0])[idx] = (_Float16)0.0f;

    const int rowbase = blockIdx.x * 64 + g * 16;

    // stage tokens for t=0
    if (sub == 0 && hi == 0)
        tkbuf[0][g][r] = tokens[(rowbase + r) * TT + 0];
    __syncthreads();

    // bias[hl][gate] : col = (gate*4 + 2*sub + hl)*16 + r
    float bias[2][4];
    #pragma unroll
    for (int hl = 0; hl < 2; ++hl)
        #pragma unroll
        for (int gt = 0; gt < 4; ++gt)
            bias[hl][gt] = b1[(gt * 4 + 2 * sub + hl) * 16 + r];

    float cst[2][8], hpv[2][8];
    #pragma unroll
    for (int a = 0; a < 2; ++a)
        #pragma unroll
        for (int j = 0; j < 8; ++j) { cst[a][j] = 0.0f; hpv[a][j] = 0.0f; }

    for (int t = 0; t < TT; ++t) {
        const int cb = t & 1;
        const int nb = (t + 1) & 1;

        // Launder zero -> fragment addresses not provably loop-invariant
        // (stops LICM from hoisting fragments into spilled registers).
        int liv = 0;
        asm volatile("" : "+v"(liv));
        const int fofs = ln * 16 + liv;

        const int tok = tkbuf[cb][g][r];

        // ---- A-frags for x from f16 embedding table (no converts) ----
        const _Float16* er = embh + (size_t)tok * EPAD;
        v16h ax[4];
        #pragma unroll
        for (int kt = 0; kt < 4; ++kt) {
            const int k0 = kt * 32 + hi * 8;
            ax[kt] = combine16(*(const half8*)(er + k0),
                               *(const half8*)(er + k0 + 16));
        }
        // ---- A-frags for h ----
        v16h ah[2];
        #pragma unroll
        for (int kt = 0; kt < 2; ++kt) {
            const int k0 = kt * 32 + hi * 8;
            ah[kt] = combine16(*(const half8*)&hbuf[cb][g][r][k0],
                               *(const half8*)&hbuf[cb][g][r][k0 + 16]);
        }
        // ---- mask per owned row ----
        float mk[8];
        #pragma unroll
        for (int j = 0; j < 8; ++j)
            mk[j] = (tkbuf[cb][g][hi * 8 + j] != 0) ? 1.0f : 0.0f;

        // ---- stage tokens for t+1 and prefetch next embedding rows ----
        if (sub == 0 && hi == 0) {
            const int tn = (t + 1 < TT) ? t + 1 : TT - 1;
            const int tknext = tokens[(rowbase + r) * TT + tn];
            tkbuf[nb][g][r] = tknext;
            const _Float16* pr = embh + (size_t)tknext * EPAD;
            __builtin_prefetch(pr, 0, 3);
            __builtin_prefetch(pr + 64, 0, 3);   // second 128B line
        }

        // ---- gates + cell update for this wave's 32 hidden cols ----
        #pragma unroll
        for (int hl = 0; hl < 2; ++hl) {
            const int ht = 2 * sub + hl;
            v8f zi = splat8(bias[hl][0]);
            v8f zf = splat8(bias[hl][1]);
            v8f zg = splat8(bias[hl][2]);
            v8f zo = splat8(bias[hl][3]);
            #pragma unroll
            for (int kt = 0; kt < 4; ++kt) {
                const _Float16* base = &Wf[(kt * 16) * 512 + fofs];
                v16h bi = *(const v16h*)(base + (ht)      * 512);
                v16h bf = *(const v16h*)(base + (ht + 4)  * 512);
                v16h bg = *(const v16h*)(base + (ht + 8)  * 512);
                v16h bo = *(const v16h*)(base + (ht + 12) * 512);
                zi = WMMA(ax[kt], bi, zi);
                zf = WMMA(ax[kt], bf, zf);
                zg = WMMA(ax[kt], bg, zg);
                zo = WMMA(ax[kt], bo, zo);
            }
            #pragma unroll
            for (int kt = 0; kt < 2; ++kt) {
                const _Float16* base = &Uf[(kt * 16) * 512 + fofs];
                v16h bi = *(const v16h*)(base + (ht)      * 512);
                v16h bf = *(const v16h*)(base + (ht + 4)  * 512);
                v16h bg = *(const v16h*)(base + (ht + 8)  * 512);
                v16h bo = *(const v16h*)(base + (ht + 12) * 512);
                zi = WMMA(ah[kt], bi, zi);
                zf = WMMA(ah[kt], bf, zf);
                zg = WMMA(ah[kt], bg, zg);
                zo = WMMA(ah[kt], bo, zo);
            }
            #pragma unroll
            for (int j = 0; j < 8; ++j) {
                float ig = sigm(zi[j]);
                float fg = sigm(zf[j]);
                float gg = tanha(zg[j]);
                float og = sigm(zo[j]);
                float cn = fg * cst[hl][j] + ig * gg;
                float hn = og * tanha(cn);
                float m  = mk[j];
                cn = m * cn + (1.0f - m) * cst[hl][j];
                hn = m * hn + (1.0f - m) * hpv[hl][j];
                cst[hl][j] = cn;
                hpv[hl][j] = hn;
                hbuf[nb][g][hi * 8 + j][ht * 16 + r] = (_Float16)hn;
            }
        }
        // ---- stream own 16x32 half to hs1[t] (intra-wave, coalesced b128) ----
        #pragma unroll
        for (int q2 = 0; q2 < 2; ++q2) {
            int c2  = ln + 32 * q2;      // 64 chunks of 16B
            int row = c2 >> 2;
            int off = (c2 & 3) * 8;      // halves
            *(uint4*)(hs1 + ((size_t)t * BB + rowbase + row) * H1 + sub * 32 + off) =
                *(const uint4*)&hbuf[nb][g][row][sub * 32 + off];
        }
        __syncthreads();
    }
}

// ----------------------------------------------------------------------------
// Layer 2: same structure; K_x=64 (hs1 f16), K_h=32, N=128, wave owns ht=sub.
// ----------------------------------------------------------------------------
__global__ __launch_bounds__(256) void lstm2_kernel(
    const int* __restrict__ tokens, const _Float16* __restrict__ hs1,
    const float* __restrict__ W2, const float* __restrict__ U2,
    const float* __restrict__ b2, float* __restrict__ hfin)
{
    __shared__ _Float16 Wf[2 * 8 * 512];
    __shared__ _Float16 Uf[1 * 8 * 512];
    __shared__ _Float16 hbuf[2][4][16][H2];
    __shared__ int      tkbuf[2][4][16];

    const int tid = threadIdx.x;
    const int wv  = tid >> 5;
    const int ln  = tid & 31;
    const int r   = ln & 15;
    const int hi  = ln >> 4;
    const int g   = wv >> 1;
    const int sub = wv & 1;

    for (int idx = tid; idx < 2 * 8 * 512; idx += 256) {
        int kt = idx >> 12;
        int nt = (idx >> 9) & 7;
        int q  = idx & 511;
        int fl = q >> 4, e = q & 15;
        int fr = fl & 15, fh = fl >> 4;
        int k = kt * 32 + fh * 8 + (e & 7) + (e >> 3) * 16;
        int n = nt * 16 + fr;
        Wf[idx] = (_Float16)W2[k * NG2 + n];
    }
    for (int idx = tid; idx < 8 * 512; idx += 256) {
        int nt = (idx >> 9) & 7;
        int q  = idx & 511;
        int fl = q >> 4, e = q & 15;
        int fr = fl & 15, fh = fl >> 4;
        int k = fh * 8 + (e & 7) + (e >> 3) * 16;
        int n = nt * 16 + fr;
        Uf[idx] = (_Float16)U2[k * NG2 + n];
    }
    for (int idx = tid; idx < 2 * 4 * 16 * H2; idx += 256)
        (&hbuf[0][0][0][0])[idx] = (_Float16)0.0f;

    const int rowbase = blockIdx.x * 64 + g * 16;

    if (sub == 0 && hi == 0)
        tkbuf[0][g][r] = tokens[(rowbase + r) * TT + 0];
    __syncthreads();

    // bias[gate] : col = (gate*2 + sub)*16 + r
    float bias[4];
    #pragma unroll
    for (int gt = 0; gt < 4; ++gt)
        bias[gt] = b2[(gt * 2 + sub) * 16 + r];

    float cst[8], hpv[8];
    #pragma unroll
    for (int j = 0; j < 8; ++j) { cst[j] = 0.0f; hpv[j] = 0.0f; }

    for (int t = 0; t < TT; ++t) {
        const int cb = t & 1;
        const int nb = (t + 1) & 1;

        int liv = 0;
        asm volatile("" : "+v"(liv));
        const int fofs = ln * 16 + liv;

        const _Float16* xr = hs1 + ((size_t)t * BB + rowbase + r) * H1;
        v16h ax[2];
        #pragma unroll
        for (int kt = 0; kt < 2; ++kt) {
            const int k0 = kt * 32 + hi * 8;
            ax[kt] = combine16(*(const half8*)(xr + k0),
                               *(const half8*)(xr + k0 + 16));
        }
        v16h ah;
        {
            const int k0 = hi * 8;
            ah = combine16(*(const half8*)&hbuf[cb][g][r][k0],
                           *(const half8*)&hbuf[cb][g][r][k0 + 16]);
        }
        float mk[8];
        #pragma unroll
        for (int j = 0; j < 8; ++j)
            mk[j] = (tkbuf[cb][g][hi * 8 + j] != 0) ? 1.0f : 0.0f;

        if (sub == 0 && hi == 0) {
            const int tn = (t + 1 < TT) ? t + 1 : TT - 1;
            tkbuf[nb][g][r] = tokens[(rowbase + r) * TT + tn];
            __builtin_prefetch(hs1 + ((size_t)tn * BB + rowbase + r) * H1, 0, 3);
        }

        const int ht = sub;
        v8f zi = splat8(bias[0]);
        v8f zf = splat8(bias[1]);
        v8f zg = splat8(bias[2]);
        v8f zo = splat8(bias[3]);
        #pragma unroll
        for (int kt = 0; kt < 2; ++kt) {
            const _Float16* base = &Wf[(kt * 8) * 512 + fofs];
            v16h bi = *(const v16h*)(base + (ht)     * 512);
            v16h bf = *(const v16h*)(base + (2 + ht) * 512);
            v16h bg = *(const v16h*)(base + (4 + ht) * 512);
            v16h bo = *(const v16h*)(base + (6 + ht) * 512);
            zi = WMMA(ax[kt], bi, zi);
            zf = WMMA(ax[kt], bf, zf);
            zg = WMMA(ax[kt], bg, zg);
            zo = WMMA(ax[kt], bo, zo);
        }
        {
            const _Float16* base = &Uf[fofs];
            v16h bi = *(const v16h*)(base + (ht)     * 512);
            v16h bf = *(const v16h*)(base + (2 + ht) * 512);
            v16h bg = *(const v16h*)(base + (4 + ht) * 512);
            v16h bo = *(const v16h*)(base + (6 + ht) * 512);
            zi = WMMA(ah, bi, zi);
            zf = WMMA(ah, bf, zf);
            zg = WMMA(ah, bg, zg);
            zo = WMMA(ah, bo, zo);
        }
        #pragma unroll
        for (int j = 0; j < 8; ++j) {
            float ig = sigm(zi[j]);
            float fg = sigm(zf[j]);
            float gg = tanha(zg[j]);
            float og = sigm(zo[j]);
            float cn = fg * cst[j] + ig * gg;
            float hn = og * tanha(cn);
            float m  = mk[j];
            cn = m * cn + (1.0f - m) * cst[j];
            hn = m * hn + (1.0f - m) * hpv[j];
            cst[j] = cn;
            hpv[j] = hn;
            hbuf[nb][g][hi * 8 + j][ht * 16 + r] = (_Float16)hn;
        }
        __syncthreads();
    }
    // final hidden -> hfin [B][H2] f32 (own 16 cols)
    #pragma unroll
    for (int j = 0; j < 8; ++j)
        hfin[(size_t)(rowbase + hi * 8 + j) * H2 + sub * 16 + r] = hpv[j];
}

// ----------------------------------------------------------------------------
// Dense head: out[b][c] = sigmoid(h[b] . Wd[:,c] + bd[c]), 1024x4
// ----------------------------------------------------------------------------
__global__ __launch_bounds__(256) void dense_kernel(
    const float* __restrict__ hfin, const float* __restrict__ Wd,
    const float* __restrict__ bd, float* __restrict__ out)
{
    int idx = blockIdx.x * 256 + threadIdx.x;
    if (idx >= BB * 4) return;
    int b = idx >> 2, c = idx & 3;
    float s = bd[c];
    #pragma unroll
    for (int h = 0; h < H2; ++h) s += hfin[b * H2 + h] * Wd[h * 4 + c];
    out[idx] = 1.0f / (1.0f + __expf(-s));
}

extern "C" void kernel_launch(void* const* d_in, const int* in_sizes, int n_in,
                              void* d_out, int out_size, void* d_ws, size_t ws_size,
                              hipStream_t stream)
{
    const int*   tokens = (const int*)d_in[0];
    const float* emb = (const float*)d_in[1];
    const float* W1  = (const float*)d_in[2];
    const float* U1  = (const float*)d_in[3];
    const float* b1  = (const float*)d_in[4];
    const float* W2  = (const float*)d_in[5];
    const float* U2  = (const float*)d_in[6];
    const float* b2  = (const float*)d_in[7];
    const float* Wd  = (const float*)d_in[8];
    const float* bd  = (const float*)d_in[9];
    float* out = (float*)d_out;

    char* ws = (char*)d_ws;
    _Float16* hs1  = (_Float16*)ws;                                   // 32 MB
    float*    hfin = (float*)(ws + (size_t)TT * BB * H1 * 2);         // 128 KB
    _Float16* embh = (_Float16*)(ws + (size_t)TT * BB * H1 * 2
                                    + (size_t)BB * H2 * 4);           // 1.28 MB

    embcvt_kernel<<<(NVOC * EPAD + 255) / 256, 256, 0, stream>>>(emb, embh);
    lstm1_kernel<<<16, 256, 0, stream>>>(tokens, embh, W1, U1, b1, hs1);
    lstm2_kernel<<<16, 256, 0, stream>>>(tokens, hs1, W2, U2, b2, hfin);
    dense_kernel<<<(BB * 4 + 255) / 256, 256, 0, stream>>>(hfin, Wd, bd, out);
}
